// ST_model_58815282151899
// MI455X (gfx1250) — matrix operand order repaired and checked by reading, not compiled, not run on hardware
//
#include <hip/hip_runtime.h>

// ---------------------------------------------------------------------------
// Types for CDNA5 WMMA (wave32): bf16 operands, f32 accumulate.
// ---------------------------------------------------------------------------
typedef __bf16 bf16_t;
typedef __bf16 bf8v  __attribute__((ext_vector_type(8)));
typedef __bf16 bf16v __attribute__((ext_vector_type(16)));
typedef float  f8v   __attribute__((ext_vector_type(8)));

#define LNEPS 1e-5f

// ---------------------------------------------------------------------------
// f32 -> bf16 cast (grid-stride)
// ---------------------------------------------------------------------------
__global__ void k_cast(const float* __restrict__ in, bf16_t* __restrict__ out, long n) {
  long i  = (long)blockIdx.x * blockDim.x + threadIdx.x;
  long st = (long)gridDim.x * blockDim.x;
  for (; i < n; i += st) out[i] = (bf16_t)in[i];
}

// ---------------------------------------------------------------------------
// acc[r, 0:64] = acc[r, 64:128] = x[r, :]   (Chebyshev T0 term, both supports)
// ---------------------------------------------------------------------------
__global__ void k_acc_init(const float* __restrict__ x, float* __restrict__ acc, long rows) {
  long n  = rows * 128;
  long i  = (long)blockIdx.x * blockDim.x + threadIdx.x;
  long st = (long)gridDim.x * blockDim.x;
  for (; i < n; i += st) {
    long r = i >> 7;
    int  c = (int)(i & 127);
    acc[i] = x[r * 64 + (c & 63)];
  }
}

// ---------------------------------------------------------------------------
// Generic batched WMMA GEMM:  res = alpha * (A @ B) + bias [- tp]  [relu]
//   then optionally: acc += res ; outF = res ; outB = bf16(res)
// Tile: 128 rows x 64 cols per block (256 thr = 8 waves; wave w -> rows w*16..).
// K stepped by 64: per barrier period each wave issues 8 x 16x16x32 WMMA.
// LDS: A tile row-major (128x64, 16KB); B tile TRANSPOSED ([col][k], 8KB) so
// B fragments are contiguous ds_load_b128.  Next tiles are prefetched
// (global_prefetch_b8) before the compute phase.
// Requires K % 64 == 0 (true for all uses: 64/128/256/1024).
// ---------------------------------------------------------------------------
__global__ __launch_bounds__(256) void k_gemm(
    const bf16_t* __restrict__ A, long lda, long sAb,
    const bf16_t* __restrict__ B, long ldb, long sBb,
    const float*  __restrict__ bias,
    const float*  __restrict__ tp,  long ldtp,  long sTpb,
    float*        __restrict__ acc, long accRS, long accCO, long sAccb,
    float*        __restrict__ outF, long ldoF, long sOFb,
    bf16_t*       __restrict__ outB, long ldoB, long sOBb,
    int K, float alpha, int relu)
{
  __shared__ bf16_t sA[128 * 64];   // 16 KB
  __shared__ bf16_t sB[64 * 64];    // 8 KB, transposed: sB[col*64 + k]

  const int  tid  = threadIdx.x, lane = tid & 31, w = tid >> 5;
  const long bz   = blockIdx.z;
  const int  r0   = blockIdx.y * 128, n0 = blockIdx.x * 64;
  const bf16_t* Ab = A + bz * sAb;
  const bf16_t* Bb = B + bz * sBb;

  f8v c0 = {}, c1 = {}, c2 = {}, c3 = {};

  for (int k0 = 0; k0 < K; k0 += 64) {
    // --- stage A tile: 128 x 64 (four 16B chunks = 32 contiguous bf16/thread)
#pragma unroll
    for (int i = 0; i < 4; i++) {
      int ci  = tid * 4 + i;
      int row = ci >> 3, ko = (ci & 7) * 8;
      *(bf8v*)&sA[row * 64 + ko] =
          *(const bf8v*)&Ab[(long)(r0 + row) * lda + (k0 + ko)];
    }
    // --- stage B tile: 64 x 64, transpose into LDS ---
#pragma unroll
    for (int i = 0; i < 2; i++) {
      int  idx = tid * 2 + i;
      int  kr = idx >> 3, cb = (idx & 7) * 8;
      bf8v tmp = *(const bf8v*)&Bb[(long)(k0 + kr) * ldb + (n0 + cb)];
#pragma unroll
      for (int j = 0; j < 8; j++) sB[(cb + j) * 64 + kr] = tmp[j];
    }
    // --- prefetch next k-step tiles into cache (global_prefetch_b8) ---
    if (k0 + 64 < K) {
      __builtin_prefetch(&Ab[(long)(r0 + (tid & 127)) * lda + (k0 + 64)], 0, 3);
      __builtin_prefetch(&Bb[(long)(k0 + 64 + (tid & 63)) * ldb + n0], 0, 3);
    }
    __syncthreads();

    const int mrow = w * 16 + (lane & 15);
    const int koA  = (lane < 16) ? 0 : 8;
    const int koB  = (lane < 16) ? 0 : 8;
#pragma unroll
    for (int kh = 0; kh < 2; kh++) {
      const int kb = kh * 32;
      // A fragment (16x32, ISA layout: lanes<16 K{0..7,16..23}, else {8..15,24..31})
      bf16v a;
      {
        bf8v lo = *(const bf8v*)&sA[mrow * 64 + kb + koA];
        bf8v hi = *(const bf8v*)&sA[mrow * 64 + kb + koA + 16];
#pragma unroll
        for (int j = 0; j < 8; j++) { a[j] = lo[j]; a[j + 8] = hi[j]; }
      }
      // 4 B fragments (32x16 each) + WMMA
#pragma unroll
      for (int jt = 0; jt < 4; jt++) {
        int  col = jt * 16 + (lane & 15);
        bf16v b;
        {
          bf8v b0 = *(const bf8v*)&sB[col * 64 + kb + koB];
          bf8v b1 = *(const bf8v*)&sB[col * 64 + kb + koB + 16];
#pragma unroll
          for (int j = 0; j < 8; j++) { b[j] = b0[j]; b[j + 8] = b1[j]; }
        }
        f8v* cp = (jt == 0) ? &c0 : (jt == 1) ? &c1 : (jt == 2) ? &c2 : &c3;
        *cp = __builtin_amdgcn_wmma_f32_16x16x32_bf16(false, a, false, b,
                                                      (short)0, *cp, false, false);
      }
    }
    __syncthreads();
  }

  // --- epilogue (C/D layout: VGPR r -> M = r + (lane<16?0:8), N = lane%16) ---
  const int nl = lane & 15;
  const int mo = (lane < 16) ? 0 : 8;
  f8v cs[4] = { c0, c1, c2, c3 };
#pragma unroll
  for (int jt = 0; jt < 4; jt++) {
    int   col = n0 + jt * 16 + nl;
    float bvv = bias ? bias[col] : 0.0f;
#pragma unroll
    for (int r = 0; r < 8; r++) {
      long  row = r0 + w * 16 + mo + r;
      float res = alpha * cs[jt][r] + bvv;
      if (tp)   res -= tp[bz * sTpb + row * ldtp + col];
      if (relu) res = fmaxf(res, 0.0f);
      if (acc)  acc[bz * sAccb + row * accRS + accCO + col] += res;
      if (outF) outF[bz * sOFb + row * ldoF + col] = res;
      if (outB) outB[bz * sOBb + row * ldoB + col] = (bf16_t)res;
    }
  }
}

// ---------------------------------------------------------------------------
// Temporal MHA per (b, n, h): T=12, hd=16. One wave per head instance.
// q/k/v are f32 in [B,T,N,D] layout; output o in bf16 (feeds Wo GEMM).
// ---------------------------------------------------------------------------
__global__ __launch_bounds__(256) void k_attn(
    const float* __restrict__ q, const float* __restrict__ k,
    const float* __restrict__ v, bf16_t* __restrict__ o)
{
  __shared__ float sq[8][12][16];
  __shared__ float sk[8][12][16];
  __shared__ float sv[8][12][16];
  const int lane = threadIdx.x & 31, w = threadIdx.x >> 5;
  const int gid  = blockIdx.x * 8 + w;        // (b*1024 + n)*4 + h
  const int h = gid & 3, bn = gid >> 2;
  const int n = bn & 1023, b = bn >> 10;

  for (int i = lane; i < 192; i += 32) {
    int  t = i >> 4, e = i & 15;
    long idx = ((long)(b * 12 + t) * 1024 + n) * 64 + h * 16 + e;
    sq[w][t][e] = q[idx]; sk[w][t][e] = k[idx]; sv[w][t][e] = v[idx];
  }
  __syncthreads();

  if (lane < 12) {
    int   t = lane;
    float s[12], mx = -3.4e38f;
#pragma unroll
    for (int tk = 0; tk < 12; tk++) {
      float d = 0.f;
#pragma unroll
      for (int e = 0; e < 16; e++) d += sq[w][t][e] * sk[w][tk][e];
      d *= 0.25f;                 // 1/sqrt(hd), hd=16
      s[tk] = d; mx = fmaxf(mx, d);
    }
    float sum = 0.f;
#pragma unroll
    for (int tk = 0; tk < 12; tk++) { s[tk] = __expf(s[tk] - mx); sum += s[tk]; }
    float inv = 1.0f / sum;
#pragma unroll
    for (int e = 0; e < 16; e++) {
      float ov = 0.f;
#pragma unroll
      for (int tk = 0; tk < 12; tk++) ov += s[tk] * sv[w][tk][e];
      long idx = ((long)(b * 12 + t) * 1024 + n) * 64 + h * 16 + e;
      o[idx] = (bf16_t)(ov * inv);
    }
  }
}

// ---------------------------------------------------------------------------
// out = LayerNorm(x + r) * g + be over D=64; one wave32 per row (2 elems/lane),
// shuffle reductions; emits f32 and bf16 copies.
// ---------------------------------------------------------------------------
__global__ __launch_bounds__(256) void k_res_ln(
    const float* __restrict__ x, const float* __restrict__ r,
    const float* __restrict__ g, const float* __restrict__ be,
    float* __restrict__ outF, bf16_t* __restrict__ outB, long rows)
{
  const int lane = threadIdx.x & 31, w = threadIdx.x >> 5;
  long row = (long)blockIdx.x * 8 + w;
  if (row >= rows) return;
  long base = row * 64;
  float v0 = x[base + lane]      + r[base + lane];
  float v1 = x[base + lane + 32] + r[base + lane + 32];
  float s = v0 + v1;
#pragma unroll
  for (int off = 16; off > 0; off >>= 1) s += __shfl_xor(s, off, 32);
  float mean = s * (1.0f / 64.0f);
  float d0 = v0 - mean, d1 = v1 - mean;
  float qv = d0 * d0 + d1 * d1;
#pragma unroll
  for (int off = 16; off > 0; off >>= 1) qv += __shfl_xor(qv, off, 32);
  float rstd = rsqrtf(qv * (1.0f / 64.0f) + LNEPS);
  float o0 = d0 * rstd * g[lane]      + be[lane];
  float o1 = d1 * rstd * g[lane + 32] + be[lane + 32];
  outF[base + lane]      = o0;
  outF[base + lane + 32] = o1;
  outB[base + lane]      = (bf16_t)o0;
  outB[base + lane + 32] = (bf16_t)o1;
}

// ---------------------------------------------------------------------------
// Output head: out[b,tau,n] = sum_{t,d} x[b,t,n,d] * Wout[t*64+d, tau] + bout
// ---------------------------------------------------------------------------
__global__ void k_outproj(const float* __restrict__ x, const float* __restrict__ Wout,
                          const float* __restrict__ bout, float* __restrict__ out)
{
  int tid = blockIdx.x * blockDim.x + threadIdx.x;   // 16*1024*12 threads
  int tau = tid % 12;
  int n   = (tid / 12) & 1023;
  int b   = tid / (12 * 1024);
  float s = bout[tau];
  for (int t = 0; t < 12; t++) {
    long xb = ((long)(b * 12 + t) * 1024 + n) * 64;
    int  wb = t * 64;
#pragma unroll 8
    for (int d = 0; d < 64; d++) s += x[xb + d] * Wout[(long)(wb + d) * 12 + tau];
  }
  out[((long)b * 12 + tau) * 1024 + n] = s;
}

// ---------------------------------------------------------------------------
extern "C" void kernel_launch(void* const* d_in, const int* in_sizes, int n_in,
                              void* d_out, int out_size, void* d_ws, size_t ws_size,
                              hipStream_t stream)
{
  (void)in_sizes; (void)n_in; (void)out_size; (void)ws_size;
  const float* x_in = (const float*)d_in[0];
  const float* A_in = (const float*)d_in[1];
  const float* Wg   = (const float*)d_in[2];
  const float* bg   = (const float*)d_in[3];
  const float* Wq   = (const float*)d_in[4];
  const float* bq   = (const float*)d_in[5];
  const float* Wk   = (const float*)d_in[6];
  const float* bk   = (const float*)d_in[7];
  const float* Wv   = (const float*)d_in[8];
  const float* bv   = (const float*)d_in[9];
  const float* Wo   = (const float*)d_in[10];
  const float* bo   = (const float*)d_in[11];
  const float* W1   = (const float*)d_in[12];
  const float* b1   = (const float*)d_in[13];
  const float* W2   = (const float*)d_in[14];
  const float* b2   = (const float*)d_in[15];
  const float* ln1g = (const float*)d_in[16];
  const float* ln1b = (const float*)d_in[17];
  const float* ln2g = (const float*)d_in[18];
  const float* ln2b = (const float*)d_in[19];
  const float* Wout = (const float*)d_in[20];
  const float* bout = (const float*)d_in[21];
  float* out = (float*)d_out;

  const long M = 16L * 12 * 1024;  // flat rows B*T*N = 196608

  // ---- workspace carve-out (~483 MB), lifetime-based aliasing ----
  char*  base = (char*)d_ws;
  size_t off  = 0;
  auto take = [&](size_t bytes) -> void* {
    void* p = base + off;
    off = (off + bytes + 255) & ~(size_t)255;
    return p;
  };
  const size_t f64  = (size_t)M * 64 * 4;
  const size_t b64s = (size_t)M * 64 * 2;
  const size_t f128 = (size_t)M * 128 * 4;
  const size_t b128 = (size_t)M * 128 * 2;

  float*  xFws = (float*)take(f64);
  bf16_t* xB   = (bf16_t*)take(b64s);
  float*  acc  = (float*)take(f128);    // later reused as f1B ([M,256] bf16, same bytes)
  bf16_t* accB = (bf16_t*)take(b128);   // later reused as x1B
  float*  t1F  = (float*)take(f64);     // later: xAttnF
  bf16_t* t1B  = (bf16_t*)take(b64s);   // later: xAttnB
  bf16_t* t2B  = (bf16_t*)take(b64s);   // later: oB
  float*  qF   = (float*)take(f64);     // later: oprojF
  float*  kFb  = (float*)take(f64);     // later: x1F
  float*  vFb  = (float*)take(f64);     // later: f2F
  bf16_t* Abf  = (bf16_t*)take((size_t)2 * 1024 * 1024 * 2);
  bf16_t* WgB  = (bf16_t*)take((size_t)3 * 128 * 64 * 2);
  bf16_t* WqB  = (bf16_t*)take((size_t)3 * 64 * 64 * 2);
  bf16_t* WkB  = (bf16_t*)take((size_t)3 * 64 * 64 * 2);
  bf16_t* WvB  = (bf16_t*)take((size_t)3 * 64 * 64 * 2);
  bf16_t* WoB  = (bf16_t*)take((size_t)3 * 64 * 64 * 2);
  bf16_t* W1B  = (bf16_t*)take((size_t)3 * 64 * 256 * 2);
  bf16_t* W2B  = (bf16_t*)take((size_t)3 * 256 * 64 * 2);

  float*  xAttnF = t1F;
  bf16_t* xAttnB = t1B;
  bf16_t* oB     = t2B;
  float*  oprojF = qF;
  float*  x1F    = kFb;
  bf16_t* x1B    = accB;
  bf16_t* f1B    = (bf16_t*)acc;
  float*  f2F    = vFb;

  auto cast = [&](const float* src, bf16_t* dst, long n) {
    int grid = (int)((n + 2047) / 2048); if (grid < 1) grid = 1;
    k_cast<<<grid, 256, 0, stream>>>(src, dst, n);
  };

  // ---- one-time bf16 conversions ----
  cast(A_in, Abf, 2L * 1024 * 1024);
  cast(Wg,  WgB, 3L * 128 * 64);
  cast(Wq,  WqB, 3L * 64 * 64);
  cast(Wk,  WkB, 3L * 64 * 64);
  cast(Wv,  WvB, 3L * 64 * 64);
  cast(Wo,  WoB, 3L * 64 * 64);
  cast(W1,  W1B, 3L * 64 * 256);
  cast(W2,  W2B, 3L * 256 * 64);
  cast(x_in, xB, M * 64);

  const float* xF = x_in;
  for (int li = 0; li < 3; li++) {
    // ================= Chebyshev GNN =================
    k_acc_init<<<4096, 256, 0, stream>>>(xF, acc, M);
    for (int s = 0; s < 2; s++) {
      const bf16_t* As = Abf + (size_t)s * 1024 * 1024;
      dim3 g(1, 8, 192);  // 64 cols, 1024 rows, 192 (b,t) batches
      // t1 = A @ x ; acc += t1
      k_gemm<<<g, 256, 0, stream>>>(As, 1024, 0,  xB, 64, 65536,
          nullptr,  nullptr, 0, 0,
          acc, 128, (long)s * 64, 131072,
          t1F, 64, 65536,  t1B, 64, 65536,
          1024, 1.0f, 0);
      // t2 = 2*A@t1 - x ; acc += t2
      k_gemm<<<g, 256, 0, stream>>>(As, 1024, 0,  t1B, 64, 65536,
          nullptr,  xF, 64, 65536,
          acc, 128, (long)s * 64, 131072,
          nullptr, 0, 0,  t2B, 64, 65536,
          1024, 2.0f, 0);
      // t3 = 2*A@t2 - t1 ; acc += t3
      k_gemm<<<g, 256, 0, stream>>>(As, 1024, 0,  t2B, 64, 65536,
          nullptr,  t1F, 64, 65536,
          acc, 128, (long)s * 64, 131072,
          nullptr, 0, 0,  nullptr, 0, 0,
          1024, 2.0f, 0);
    }
    cast(acc, accB, M * 128);
    // x_attn = relu(acc @ Wg + bg)
    k_gemm<<<dim3(1, 1536, 1), 256, 0, stream>>>(accB, 128, 0,
        WgB + (size_t)li * 128 * 64, 64, 0,
        bg + li * 64,  nullptr, 0, 0,  nullptr, 0, 0, 0,
        xAttnF, 64, 0,  xAttnB, 64, 0,
        128, 1.0f, 1);

    // ================= temporal MHA =================
    k_gemm<<<dim3(1, 1536, 1), 256, 0, stream>>>(xAttnB, 64, 0,
        WqB + (size_t)li * 64 * 64, 64, 0, bq + li * 64,
        nullptr, 0, 0,  nullptr, 0, 0, 0,
        qF, 64, 0,  nullptr, 0, 0,  64, 1.0f, 0);
    k_gemm<<<dim3(1, 1536, 1), 256, 0, stream>>>(xAttnB, 64, 0,
        WkB + (size_t)li * 64 * 64, 64, 0, bk + li * 64,
        nullptr, 0, 0,  nullptr, 0, 0, 0,
        kFb, 64, 0,  nullptr, 0, 0,  64, 1.0f, 0);
    k_gemm<<<dim3(1, 1536, 1), 256, 0, stream>>>(xAttnB, 64, 0,
        WvB + (size_t)li * 64 * 64, 64, 0, bv + li * 64,
        nullptr, 0, 0,  nullptr, 0, 0, 0,
        vFb, 64, 0,  nullptr, 0, 0,  64, 1.0f, 0);
    k_attn<<<8192, 256, 0, stream>>>(qF, kFb, vFb, oB);
    k_gemm<<<dim3(1, 1536, 1), 256, 0, stream>>>(oB, 64, 0,
        WoB + (size_t)li * 64 * 64, 64, 0, bo + li * 64,
        nullptr, 0, 0,  nullptr, 0, 0, 0,
        oprojF, 64, 0,  nullptr, 0, 0,  64, 1.0f, 0);
    k_res_ln<<<24576, 256, 0, stream>>>(xAttnF, oprojF,
        ln1g + li * 64, ln1b + li * 64, x1F, x1B, M);

    // ================= FFN =================
    k_gemm<<<dim3(4, 1536, 1), 256, 0, stream>>>(x1B, 64, 0,
        W1B + (size_t)li * 64 * 256, 256, 0, b1 + li * 256,
        nullptr, 0, 0,  nullptr, 0, 0, 0,
        nullptr, 0, 0,  f1B, 256, 0,  64, 1.0f, 1);
    k_gemm<<<dim3(1, 1536, 1), 256, 0, stream>>>(f1B, 256, 0,
        W2B + (size_t)li * 256 * 64, 64, 0, b2 + li * 64,
        nullptr, 0, 0,  nullptr, 0, 0, 0,
        f2F, 64, 0,  nullptr, 0, 0,  256, 1.0f, 0);
    k_res_ln<<<24576, 256, 0, stream>>>(x1F, f2F,
        ln2g + li * 64, ln2b + li * 64, xFws, xB, M);
    xF = xFws;
  }

  // ================= output head =================
  k_outproj<<<768, 256, 0, stream>>>(xFws, Wout, bout, out);
}
